// New_27857157882089
// MI455X (gfx1250) — matrix operand order repaired and checked by reading, not compiled, not run on hardware
//
#include <hip/hip_runtime.h>
#include <math.h>

// ---------------- problem constants ----------------
#define Bv 1024
#define Sv 32
#define Cv 256
#define Hv 8
#define Fv 128
#define Nv 50000
#define Ev 400000
#define EPSv 1e-5f
#define AVGLOG 2.1972245773362196f   // log(9)

typedef __attribute__((ext_vector_type(16))) __bf16 v16bf;
typedef __attribute__((ext_vector_type(8)))  float  v8f;
typedef __attribute__((ext_vector_type(2)))  float  v2f;
typedef __attribute__((ext_vector_type(2)))  __bf16 v2bf;

union Frag { v16bf v; unsigned u[8]; };

// f32 pair -> packed bf16 dword with *forced* operand pairing:
// fptrunc <2 x float> -> <2 x bfloat> selects one v_cvt_pk_bf16_f32.
__device__ __forceinline__ unsigned cvt2(float x, float y) {
    v2f f; f.x = x; f.y = y;
    v2bf h = __builtin_convertvector(f, v2bf);
    unsigned u;
    __builtin_memcpy(&u, &h, 4);
    return u;
}
__device__ __forceinline__ unsigned short bf1(float x) {
    union { __bf16 h; unsigned short u; } t; t.h = (__bf16)x; return t.u;
}
// order-preserving float<->uint encoding for atomic max/min
__device__ __forceinline__ unsigned fenc(float f) {
    union { float f; unsigned u; } a; a.f = f;
    return (a.u >> 31) ? ~a.u : (a.u | 0x80000000u);
}
__device__ __forceinline__ float fdec(unsigned k) {
    union { float f; unsigned u; } a;
    a.u = (k >> 31) ? (k & 0x7FFFFFFFu) : ~k;
    return a.f;
}

// A fragment (16x32 bf16, ISA layout) from fp32 row chunk p[0..31] via float4 loads
__device__ __forceinline__ void loadA(Frag& a, const float* __restrict__ p, int half) {
    const float4* q = (const float4*)p;
    float4 l0 = q[half * 2], l1 = q[half * 2 + 1];
    float4 h0 = q[half * 2 + 4], h1 = q[half * 2 + 5];
    a.u[0] = cvt2(l0.x, l0.y); a.u[1] = cvt2(l0.z, l0.w);
    a.u[2] = cvt2(l1.x, l1.y); a.u[3] = cvt2(l1.z, l1.w);
    a.u[4] = cvt2(h0.x, h0.y); a.u[5] = cvt2(h0.z, h0.w);
    a.u[6] = cvt2(h1.x, h1.y); a.u[7] = cvt2(h1.z, h1.w);
}
__device__ __forceinline__ void loadA_scaled(Frag& a, const float* __restrict__ p, int half, float sc) {
    const float4* q = (const float4*)p;
    float4 l0 = q[half * 2], l1 = q[half * 2 + 1];
    float4 h0 = q[half * 2 + 4], h1 = q[half * 2 + 5];
    a.u[0] = cvt2(l0.x * sc, l0.y * sc); a.u[1] = cvt2(l0.z * sc, l0.w * sc);
    a.u[2] = cvt2(l1.x * sc, l1.y * sc); a.u[3] = cvt2(l1.z * sc, l1.w * sc);
    a.u[4] = cvt2(h0.x * sc, h0.y * sc); a.u[5] = cvt2(h0.z * sc, h0.w * sc);
    a.u[6] = cvt2(h1.x * sc, h1.y * sc); a.u[7] = cvt2(h1.z * sc, h1.w * sc);
}
// B fragment (32x16 bf16) from tile-contiguous packed weight:
// layout Wp[((k0/32)*N + col)*16 + pp], pp = half*4 + j (j<4), 8 + half*4 + (j-4) (j>=4)  ->  2x b128 loads
__device__ __forceinline__ void loadB(Frag& b, const unsigned* __restrict__ Wp,
                                      int k0, int N, int col, int half) {
    const uint4* p = (const uint4*)(Wp + ((size_t)(k0 >> 5) * N + col) * 16 + half * 4);
    uint4 q0 = p[0], q1 = p[2];
    b.u[0] = q0.x; b.u[1] = q0.y; b.u[2] = q0.z; b.u[3] = q0.w;
    b.u[4] = q1.x; b.u[5] = q1.y; b.u[6] = q1.z; b.u[7] = q1.w;
}

// ===== pack fp32 weight [K,N] -> bf16 tile-contiguous layout [(K/32),N,16] dwords =====
__global__ void pack_w_k(const float* __restrict__ W, unsigned* __restrict__ Wp,
                         int K2, int N) {   // K2 = K/2 (total pair count per column)
    int idx = blockIdx.x * 256 + threadIdx.x;
    if (idx >= K2 * N) return;
    int pp = idx & 15;
    int tmp = idx >> 4;
    int col = tmp % N;
    int tile = tmp / N;
    int k = tile * 32 + 2 * pp;
    Wp[idx] = cvt2(W[(size_t)k * N + col], W[(size_t)(k + 1) * N + col]);
}

// ============ generic GEMM: out = act(A[M,K] @ W[K,N] + bias) ============
// grid (M/16, N/16), block 32 (one wave per 16x16 tile); bf16 WMMA, K%32==0
template <bool RELU>
__global__ void gemm_bias_k(const float* __restrict__ A, const unsigned* __restrict__ Wp,
                            const float* __restrict__ bias, float* __restrict__ C,
                            int M, int N, int K) {
    const int lane = threadIdx.x & 31;
    const int r = lane & 15, half = lane >> 4;
    const int m0 = blockIdx.x * 16, n0 = blockIdx.y * 16;
    const float* arow = A + (size_t)(m0 + r) * K;
    const int col = n0 + r;
    v8f acc = {0.f,0.f,0.f,0.f,0.f,0.f,0.f,0.f};
    for (int k0 = 0; k0 < K; k0 += 32) {
        Frag a, b;
        loadA(a, arow + k0, half);
        loadB(b, Wp, k0, N, col, half);
        acc = __builtin_amdgcn_wmma_f32_16x16x32_bf16(false, a.v, false, b.v,
                                                      (short)0, acc, false, false);
    }
    float bv = bias[col];
#pragma unroll
    for (int rr = 0; rr < 8; ++rr) {
        float o = acc[rr] + bv;
        if (RELU) o = fmaxf(o, 0.f);
        C[(size_t)(m0 + rr + 8 * half) * N + col] = o;
    }
}

// ============ attention: one wave per (batch,head), online softmax ============
__global__ void attn_k(const float* __restrict__ q, const float* __restrict__ k,
                       const float* __restrict__ v, float* __restrict__ ao) {
    int w = (blockIdx.x * blockDim.x + threadIdx.x) >> 5;  // 0..B*H-1
    int s = threadIdx.x & 31;                              // query position
    int b = w >> 3, h = w & 7;
    const float* qrow = q + ((size_t)(b * Sv + s)) * Cv + h * 32;
    float qv[32], out[32];
#pragma unroll
    for (int d = 0; d < 32; ++d) { qv[d] = qrow[d]; out[d] = 0.f; }
    float mrun = -3.4e38f, lrun = 0.f;
    for (int t = 0; t < 32; ++t) {
        const float* kr = k + ((size_t)(b * Sv + t)) * Cv + h * 32;
        float sc = 0.f;
#pragma unroll
        for (int d = 0; d < 32; ++d) sc += qv[d] * kr[d];
        sc *= 0.17677669529663687f;  // 1/sqrt(32)
        float mn = fmaxf(mrun, sc);
        float c = __expf(mrun - mn), p = __expf(sc - mn);
        lrun = lrun * c + p;
        const float* vr = v + ((size_t)(b * Sv + t)) * Cv + h * 32;
#pragma unroll
        for (int d = 0; d < 32; ++d) out[d] = out[d] * c + p * vr[d];
        mrun = mn;
    }
    float inv = 1.f / lrun;
    float* orow = ao + ((size_t)(b * Sv + s)) * Cv + h * 32;
#pragma unroll
    for (int d = 0; d < 32; ++d) orow[d] = out[d] * inv;
}

// ============ out = LN(x + y)  (row length 256, one wave per row) ============
__global__ void add_ln_k(const float* __restrict__ x, const float* __restrict__ y,
                         const float* __restrict__ g, const float* __restrict__ b,
                         float* __restrict__ out) {
    int row = (blockIdx.x * blockDim.x + threadIdx.x) >> 5;
    int lane = threadIdx.x & 31;
    const float* xr = x + (size_t)row * Cv;
    const float* yr = y + (size_t)row * Cv;
    float v[8]; float s = 0.f;
#pragma unroll
    for (int i = 0; i < 8; ++i) { v[i] = xr[lane + 32 * i] + yr[lane + 32 * i]; s += v[i]; }
#pragma unroll
    for (int off = 16; off; off >>= 1) s += __shfl_xor(s, off, 32);
    float mu = s * (1.f / Cv), qv = 0.f;
#pragma unroll
    for (int i = 0; i < 8; ++i) { float d = v[i] - mu; qv += d * d; }
#pragma unroll
    for (int off = 16; off; off >>= 1) qv += __shfl_xor(qv, off, 32);
    float rs = rsqrtf(qv * (1.f / Cv) + EPSv);
    float* orow = out + (size_t)row * Cv;
#pragma unroll
    for (int i = 0; i < 8; ++i) {
        int c = lane + 32 * i;
        orow[c] = (v[i] - mu) * rs * g[c] + b[c];
    }
}

// ============ out = LN3(LN2(x + f))  fused double layernorm ============
__global__ void add_ln2_k(const float* __restrict__ x, const float* __restrict__ f,
                          const float* __restrict__ g2, const float* __restrict__ b2,
                          const float* __restrict__ g3, const float* __restrict__ b3,
                          float* __restrict__ out) {
    int row = (blockIdx.x * blockDim.x + threadIdx.x) >> 5;
    int lane = threadIdx.x & 31;
    const float* xr = x + (size_t)row * Cv;
    const float* fr = f + (size_t)row * Cv;
    float v[8]; float s = 0.f;
#pragma unroll
    for (int i = 0; i < 8; ++i) { v[i] = xr[lane + 32 * i] + fr[lane + 32 * i]; s += v[i]; }
#pragma unroll
    for (int off = 16; off; off >>= 1) s += __shfl_xor(s, off, 32);
    float mu = s * (1.f / Cv), qv = 0.f;
#pragma unroll
    for (int i = 0; i < 8; ++i) { float d = v[i] - mu; qv += d * d; }
#pragma unroll
    for (int off = 16; off; off >>= 1) qv += __shfl_xor(qv, off, 32);
    float rs = rsqrtf(qv * (1.f / Cv) + EPSv);
    float t[8]; s = 0.f;
#pragma unroll
    for (int i = 0; i < 8; ++i) {
        int c = lane + 32 * i;
        t[i] = (v[i] - mu) * rs * g2[c] + b2[c];
        s += t[i];
    }
#pragma unroll
    for (int off = 16; off; off >>= 1) s += __shfl_xor(s, off, 32);
    mu = s * (1.f / Cv); qv = 0.f;
#pragma unroll
    for (int i = 0; i < 8; ++i) { float d = t[i] - mu; qv += d * d; }
#pragma unroll
    for (int off = 16; off; off >>= 1) qv += __shfl_xor(qv, off, 32);
    rs = rsqrtf(qv * (1.f / Cv) + EPSv);
    float* orow = out + (size_t)row * Cv;
#pragma unroll
    for (int i = 0; i < 8; ++i) {
        int c = lane + 32 * i;
        orow[c] = (t[i] - mu) * rs * g3[c] + b3[c];
    }
}

// ============ fold edge-encoder into pre-NN weight (Wm[384,128], bm[128]) ====
__global__ void prep_wm_k(const float* __restrict__ We, const float* __restrict__ be,
                          const float* __restrict__ Wpre, const float* __restrict__ bpre,
                          float* __restrict__ Wm, float* __restrict__ bm) {
    int idx = blockIdx.x * 256 + threadIdx.x;
    if (idx < 384 * 128) {
        int i = idx >> 7, j = idx & 127;
        if (i < 256) { Wm[idx] = Wpre[idx]; }
        else {
            int kk = i - 256; float s = 0.f;
            for (int t = 0; t < 128; ++t) s += We[kk * 128 + t] * Wpre[(256 + t) * 128 + j];
            Wm[idx] = s;
        }
    } else if (idx < 384 * 128 + 128) {
        int j = idx - 384 * 128; float s = bpre[j];
        for (int t = 0; t < 128; ++t) s += be[t] * Wpre[(256 + t) * 128 + j];
        bm[j] = s;
    }
}

// ============ fold Wpost @ Wlin (Wpl[1664,128], bpl[128]) ============
__global__ void prep_wpost_k(const float* __restrict__ Wpost, const float* __restrict__ bpost,
                             const float* __restrict__ Wlin, const float* __restrict__ blin,
                             float* __restrict__ Wpl, float* __restrict__ bpl) {
    int idx = blockIdx.x * 256 + threadIdx.x;
    if (idx < 1664 * 128) {
        int i = idx >> 7, j = idx & 127; float s = 0.f;
        for (int t = 0; t < 128; ++t) s += Wpost[(size_t)i * 128 + t] * Wlin[t * 128 + j];
        Wpl[idx] = s;
    } else if (idx < 1664 * 128 + 128) {
        int j = idx - 1664 * 128; float s = blin[j];
        for (int t = 0; t < 128; ++t) s += bpost[t] * Wlin[t * 128 + j];
        bpl[j] = s;
    }
}

// ============ PNA pre-NN gather-GEMM: m = [x[dst]|x[src]|ea] @ Wm + bm ======
// grid (E/16, 128/16), block 32
__global__ void pna_pre_k(const float* __restrict__ xg, const float* __restrict__ ea,
                          const int* __restrict__ src, const int* __restrict__ dst,
                          const unsigned* __restrict__ Wmp, const float* __restrict__ bm,
                          float* __restrict__ Mout) {
    const int lane = threadIdx.x & 31;
    const int r = lane & 15, half = lane >> 4;
    const int e0 = blockIdx.x * 16, n0 = blockIdx.y * 16;
    const int e = e0 + r;
    const float* xd = xg + (size_t)dst[e] * Fv;
    const float* xs = xg + (size_t)src[e] * Fv;
    const float* er = ea + (size_t)e * Fv;
    const int col = n0 + r;
    v8f acc = {0.f,0.f,0.f,0.f,0.f,0.f,0.f,0.f};
    for (int k0 = 0; k0 < 384; k0 += 32) {
        const float* sp = (k0 < 128) ? xd : (k0 < 256) ? xs : er;
        Frag a, b;
        loadA(a, sp + (k0 & 127), half);
        loadB(b, Wmp, k0, Fv, col, half);
        acc = __builtin_amdgcn_wmma_f32_16x16x32_bf16(false, a.v, false, b.v,
                                                      (short)0, acc, false, false);
    }
    float bv = bm[col];
#pragma unroll
    for (int rr = 0; rr < 8; ++rr)
        Mout[(size_t)(e0 + rr + 8 * half) * Fv + col] = acc[rr] + bv;
}

// ============ aggregation buffers init / scatter / finalize ============
__global__ void aggr_init_k(float* sum, float* sum2, unsigned* mx, unsigned* mn, float* cnt) {
    size_t idx = (size_t)blockIdx.x * 256 + threadIdx.x;
    if (idx < (size_t)Nv * Fv) {
        sum[idx] = 0.f; sum2[idx] = 0.f;
        mx[idx] = 0x007FFFFFu;   // fenc(-inf)
        mn[idx] = 0xFF800000u;   // fenc(+inf)
    }
    if (idx < Nv) cnt[idx] = 0.f;
}

__global__ void pna_scatter_k(const float* __restrict__ m, const int* __restrict__ dst,
                              float* sum, float* sum2, unsigned* mx, unsigned* mn, float* cnt) {
    size_t idx = (size_t)blockIdx.x * 256 + threadIdx.x;
    if (idx >= (size_t)Ev * Fv) return;
    int e = (int)(idx >> 7), j = (int)(idx & 127);
    int d = dst[e];
    float v = m[idx];
    size_t o = (size_t)d * Fv + j;
    atomicAdd(&sum[o], v);
    atomicAdd(&sum2[o], v * v);
    unsigned k = fenc(v);
    atomicMax(&mx[o], k);
    atomicMin(&mn[o], k);
    if (j == 0) atomicAdd(&cnt[d], 1.f);
}

__global__ void pna_final_k(float* sum, float* sum2, unsigned* mx, unsigned* mn,
                            const float* __restrict__ cnt, float* __restrict__ ld) {
    size_t idx = (size_t)blockIdx.x * 256 + threadIdx.x;
    if (idx >= (size_t)Nv * Fv) return;
    int n = (int)(idx >> 7), j = (int)(idx & 127);
    float c = cnt[n], denom = fmaxf(c, 1.f);
    float mean = sum[idx] / denom;
    float m2 = sum2[idx] / denom;
    float sd = sqrtf(fmaxf(m2 - mean * mean, 0.f) + EPSv);
    bool has = c > 0.f;
    float mxv = has ? fdec(mx[idx]) : 0.f;
    float mnv = has ? fdec(mn[idx]) : 0.f;
    sum[idx] = mean; sum2[idx] = sd;
    ((float*)mx)[idx] = mxv; ((float*)mn)[idx] = mnv;
    if (j == 0) ld[n] = logf(denom + 1.f);
}

// ============ PNA post GEMM: conv = [x | agg*scalers](N,1664) @ Wpl + bpl ====
// grid (N/16, 8), block 32
__global__ void pna_post_k(const float* __restrict__ xg, const float* __restrict__ meanb,
                           const float* __restrict__ mxb, const float* __restrict__ mnb,
                           const float* __restrict__ stdb, const float* __restrict__ ldb,
                           const unsigned* __restrict__ Wp, const float* __restrict__ bias,
                           float* __restrict__ conv) {
    const int lane = threadIdx.x & 31;
    const int r = lane & 15, half = lane >> 4;
    const int m0 = blockIdx.x * 16, n0 = blockIdx.y * 16;
    const int n = m0 + r;
    const float* xr = xg + (size_t)n * Fv;
    const float* aggp[4] = { meanb + (size_t)n * Fv, mxb + (size_t)n * Fv,
                             mnb + (size_t)n * Fv, stdb + (size_t)n * Fv };
    float ld = ldb[n];
    float sgrp[3] = { 1.f, ld * (1.f / AVGLOG), AVGLOG / ld };
    const int col = n0 + r;
    v8f acc = {0.f,0.f,0.f,0.f,0.f,0.f,0.f,0.f};
    for (int k0 = 0; k0 < 1664; k0 += 32) {
        const float* sp; float sc;
        if (k0 < 128) { sp = xr; sc = 1.f; }
        else {
            int t = k0 - 128;
            sp = aggp[(t >> 7) & 3];
            sc = sgrp[t >> 9];
        }
        Frag a, b;
        loadA_scaled(a, sp + (k0 & 127), half, sc);
        loadB(b, Wp, k0, Fv, col, half);
        acc = __builtin_amdgcn_wmma_f32_16x16x32_bf16(false, a.v, false, b.v,
                                                      (short)0, acc, false, false);
    }
    float bv = bias[col];
#pragma unroll
    for (int rr = 0; rr < 8; ++rr)
        conv[(size_t)(m0 + rr + 8 * half) * Fv + col] = acc[rr] + bv;
}

// ============ batchnorm stats over N rows (one block per feature) ============
__global__ void bn_stats_k(const float* __restrict__ conv, float* bnm, float* bnrv) {
    int j = blockIdx.x, tid = threadIdx.x;
    float s = 0.f, s2 = 0.f;
    for (int i = tid; i < Nv; i += 256) {
        float v = conv[(size_t)i * Fv + j];
        s += v; s2 += v * v;
    }
    __shared__ float shs[256], shq[256];
    shs[tid] = s; shq[tid] = s2;
    __syncthreads();
    for (int off = 128; off; off >>= 1) {
        if (tid < off) { shs[tid] += shs[tid + off]; shq[tid] += shq[tid + off]; }
        __syncthreads();
    }
    if (tid == 0) {
        float mu = shs[0] * (1.f / Nv);
        float var = shq[0] * (1.f / Nv) - mu * mu;
        bnm[j] = mu;
        bnrv[j] = rsqrtf(var + EPSv);
    }
}

__global__ void gnn_update_k(const float* __restrict__ xg, const float* __restrict__ conv,
                             const float* __restrict__ bnm, const float* __restrict__ bnrv,
                             const float* __restrict__ g, const float* __restrict__ b,
                             float* __restrict__ out) {
    size_t idx = (size_t)blockIdx.x * 256 + threadIdx.x;
    if (idx >= (size_t)Nv * Fv) return;
    int j = (int)(idx & 127);
    float t = (conv[idx] - bnm[j]) * bnrv[j] * g[j] + b[j];
    out[idx] = (xg[idx] + fmaxf(t, 0.f)) * 0.5f;
}

// ============ fused 2-layer edge-update MLP (one wave per 16 edges) ============
// h = relu([xg[src]|xg[dst]|ea] @ Weu1 + beu1); out_ea = ea + (h @ Weu2 + beu2)/2
__global__ void edge_mlp_k(const float* __restrict__ xg, const float* __restrict__ ea,
                           const int* __restrict__ src, const int* __restrict__ dst,
                           const unsigned* __restrict__ W1p, const float* __restrict__ b1,
                           const unsigned* __restrict__ W2p, const float* __restrict__ b2,
                           float* __restrict__ out) {
    __shared__ unsigned short hshared[16 * Fv];  // 4 KB bf16 hidden tile
    const int lane = threadIdx.x & 31;
    const int r = lane & 15, half = lane >> 4;
    const int e0 = blockIdx.x * 16;
    const int e = e0 + r;
    const float* xs = xg + (size_t)src[e] * Fv;
    const float* xd = xg + (size_t)dst[e] * Fv;
    const float* er = ea + (size_t)e * Fv;
    v8f acc[8];
#pragma unroll
    for (int nt = 0; nt < 8; ++nt) acc[nt] = (v8f){0.f,0.f,0.f,0.f,0.f,0.f,0.f,0.f};
    for (int k0 = 0; k0 < 384; k0 += 32) {
        const float* sp = (k0 < 128) ? xs : (k0 < 256) ? xd : er;
        Frag a;
        loadA(a, sp + (k0 & 127), half);
#pragma unroll
        for (int nt = 0; nt < 8; ++nt) {
            Frag b;
            loadB(b, W1p, k0, Fv, nt * 16 + r, half);
            acc[nt] = __builtin_amdgcn_wmma_f32_16x16x32_bf16(false, a.v, false, b.v,
                                                              (short)0, acc[nt], false, false);
        }
    }
    // bias + relu, C-layout -> LDS (row-major bf16 [16][128])
#pragma unroll
    for (int nt = 0; nt < 8; ++nt) {
        int col = nt * 16 + r;
        float bv = b1[col];
#pragma unroll
        for (int rr = 0; rr < 8; ++rr) {
            float h = fmaxf(acc[nt][rr] + bv, 0.f);
            hshared[(rr + 8 * half) * Fv + col] = bf1(h);
        }
    }
    __syncthreads();
    const unsigned* h32 = (const unsigned*)hshared;
    // second layer: K = 128 (4 WMMA steps) x 8 output tiles
#pragma unroll
    for (int nt = 0; nt < 8; ++nt) {
        int col = nt * 16 + r;
        v8f acc2 = {0.f,0.f,0.f,0.f,0.f,0.f,0.f,0.f};
#pragma unroll
        for (int ks = 0; ks < 4; ++ks) {
            Frag a2, b2;
#pragma unroll
            for (int j = 0; j < 4; ++j) {
                a2.u[j]     = h32[r * 64 + ks * 16 + half * 4 + j];
                a2.u[j + 4] = h32[r * 64 + ks * 16 + 8 + half * 4 + j];
            }
            loadB(b2, W2p, ks * 32, Fv, col, half);
            acc2 = __builtin_amdgcn_wmma_f32_16x16x32_bf16(false, a2.v, false, b2.v,
                                                           (short)0, acc2, false, false);
        }
        float bv = b2[col];
#pragma unroll
        for (int rr = 0; rr < 8; ++rr) {
            size_t o = (size_t)(e0 + rr + 8 * half) * Fv + col;
            out[o] = ea[o] + (acc2[rr] + bv) * 0.5f;
        }
    }
}

// =========================== host launch ===========================
extern "C" void kernel_launch(void* const* d_in, const int* in_sizes, int n_in,
                              void* d_out, int out_size, void* d_ws, size_t ws_size,
                              hipStream_t stream) {
    (void)in_sizes; (void)n_in; (void)out_size; (void)ws_size;
    const float* x_tab = (const float*)d_in[0];
    const float* x_gnn = (const float*)d_in[1];
    const int*   ei    = (const int*)d_in[2];
    const float* ea    = (const float*)d_in[3];
    const float* Wq = (const float*)d_in[4],  *Wk = (const float*)d_in[5];
    const float* Wv = (const float*)d_in[6],  *Wo = (const float*)d_in[7];
    const float* bq = (const float*)d_in[8],  *bk = (const float*)d_in[9];
    const float* bv_ = (const float*)d_in[10], *bo = (const float*)d_in[11];
    const float* W1 = (const float*)d_in[12], *b1 = (const float*)d_in[13];
    const float* W2 = (const float*)d_in[14], *b2 = (const float*)d_in[15];
    const float* ln1g = (const float*)d_in[16], *ln2g = (const float*)d_in[17];
    const float* tabng = (const float*)d_in[18];
    const float* ln1b = (const float*)d_in[19], *ln2b = (const float*)d_in[20];
    const float* tabnb = (const float*)d_in[21];
    const float* We = (const float*)d_in[22], *be = (const float*)d_in[23];
    const float* Wpre = (const float*)d_in[24], *bpre = (const float*)d_in[25];
    const float* Wpost = (const float*)d_in[26], *bpost = (const float*)d_in[27];
    const float* Wlin = (const float*)d_in[28], *blin = (const float*)d_in[29];
    const float* bng = (const float*)d_in[30], *bnb = (const float*)d_in[31];
    const float* Weu1 = (const float*)d_in[32], *beu1 = (const float*)d_in[33];
    const float* Weu2 = (const float*)d_in[34], *beu2 = (const float*)d_in[35];
    const int* src = ei;           // edge_index[0]
    const int* dst = ei + Ev;      // edge_index[1]

    float* out_tab = (float*)d_out;                        // [B*S*C]
    float* out_xg  = out_tab + (size_t)Bv * Sv * Cv;       // [N*F]
    float* out_ea  = out_xg + (size_t)Nv * Fv;             // [E*F]

    const size_t T = (size_t)Bv * Sv * Cv;      // 8,388,608
    const size_t NF = (size_t)Nv * Fv;          // 6,400,000
    const size_t EF = (size_t)Ev * Fv;          // 51,200,000
    float* base = (float*)d_ws;
    float* qb = base;            // tab scratch (inside region A)
    float* kb = base + T;
    float* vb = base + 2 * T;
    float* aob = base + 3 * T;
    float* mb = base;            // [E,F] message tensor reuses region A (tab dead by then)
    size_t cur = EF;
    float* sumb = base + cur;  cur += NF;
    float* sum2b = base + cur; cur += NF;
    unsigned* mxb = (unsigned*)(base + cur); cur += NF;
    unsigned* mnb = (unsigned*)(base + cur); cur += NF;
    float* cntb = base + cur; cur += 50048;
    float* ldb  = base + cur; cur += 50048;
    float* convb = base + cur; cur += NF;
    float* Wmb = base + cur; cur += 384 * 128;
    float* bmb = base + cur; cur += 128;
    float* Wplb = base + cur; cur += 1664 * 128;
    float* bplb = base + cur; cur += 128;
    float* bnm = base + cur; cur += 128;
    float* bnrv = base + cur; cur += 128;
    // packed bf16 tile-contiguous weights (dwords)
    unsigned* Wqp = (unsigned*)(base + cur); cur += 128 * 256;
    unsigned* Wkp = (unsigned*)(base + cur); cur += 128 * 256;
    unsigned* Wvp = (unsigned*)(base + cur); cur += 128 * 256;
    unsigned* Wop = (unsigned*)(base + cur); cur += 128 * 256;
    unsigned* W1p = (unsigned*)(base + cur); cur += 128 * 256;
    unsigned* W2p = (unsigned*)(base + cur); cur += 128 * 256;
    unsigned* Wmp = (unsigned*)(base + cur); cur += 192 * 128;
    unsigned* Wplp = (unsigned*)(base + cur); cur += 832 * 128;
    unsigned* Weu1p = (unsigned*)(base + cur); cur += 192 * 128;
    unsigned* Weu2p = (unsigned*)(base + cur); cur += 64 * 128;

    const int MT = Bv * Sv;  // 32768 tab rows

    // --- weight folding + bf16 pre-packing ---
    prep_wm_k<<<(384 * 128 + 128 + 255) / 256, 256, 0, stream>>>(We, be, Wpre, bpre, Wmb, bmb);
    prep_wpost_k<<<(1664 * 128 + 128 + 255) / 256, 256, 0, stream>>>(Wpost, bpost, Wlin, blin, Wplb, bplb);
    const int PB = (128 * 256 + 255) / 256;
    pack_w_k<<<PB, 256, 0, stream>>>(Wq, Wqp, 128, 256);
    pack_w_k<<<PB, 256, 0, stream>>>(Wk, Wkp, 128, 256);
    pack_w_k<<<PB, 256, 0, stream>>>(Wv, Wvp, 128, 256);
    pack_w_k<<<PB, 256, 0, stream>>>(Wo, Wop, 128, 256);
    pack_w_k<<<PB, 256, 0, stream>>>(W1, W1p, 128, 256);
    pack_w_k<<<PB, 256, 0, stream>>>(W2, W2p, 128, 256);
    pack_w_k<<<(192 * 128 + 255) / 256, 256, 0, stream>>>(Wmb, Wmp, 192, 128);
    pack_w_k<<<(832 * 128 + 255) / 256, 256, 0, stream>>>(Wplb, Wplp, 832, 128);
    pack_w_k<<<(192 * 128 + 255) / 256, 256, 0, stream>>>(Weu1, Weu1p, 192, 128);
    pack_w_k<<<(64 * 128 + 255) / 256, 256, 0, stream>>>(Weu2, Weu2p, 64, 128);

    // --- tabular transformer layer ---
    dim3 gT(MT / 16, Cv / 16);
    gemm_bias_k<false><<<gT, 32, 0, stream>>>(x_tab, Wqp, bq, qb, MT, Cv, Cv);
    gemm_bias_k<false><<<gT, 32, 0, stream>>>(x_tab, Wkp, bk, kb, MT, Cv, Cv);
    gemm_bias_k<false><<<gT, 32, 0, stream>>>(x_tab, Wvp, bv_, vb, MT, Cv, Cv);
    attn_k<<<(Bv * Hv) / 8, 256, 0, stream>>>(qb, kb, vb, aob);
    gemm_bias_k<false><<<gT, 32, 0, stream>>>(aob, Wop, bo, qb, MT, Cv, Cv);   // o-proj -> qb
    add_ln_k<<<MT / 8, 256, 0, stream>>>(x_tab, qb, ln1g, ln1b, kb);           // x1 -> kb
    gemm_bias_k<true><<<gT, 32, 0, stream>>>(kb, W1p, b1, qb, MT, Cv, Cv);     // relu hidden -> qb
    gemm_bias_k<false><<<gT, 32, 0, stream>>>(qb, W2p, b2, vb, MT, Cv, Cv);    // ffn out -> vb
    add_ln2_k<<<MT / 8, 256, 0, stream>>>(kb, vb, ln2g, ln2b, tabng, tabnb, out_tab);

    // --- PNA conv ---
    dim3 gM(Ev / 16, Fv / 16);
    pna_pre_k<<<gM, 32, 0, stream>>>(x_gnn, ea, src, dst, Wmp, bmb, mb);
    aggr_init_k<<<(unsigned)((NF + 255) / 256), 256, 0, stream>>>(sumb, sum2b, mxb, mnb, cntb);
    pna_scatter_k<<<(unsigned)((EF + 255) / 256), 256, 0, stream>>>(mb, dst, sumb, sum2b, mxb, mnb, cntb);
    pna_final_k<<<(unsigned)((NF + 255) / 256), 256, 0, stream>>>(sumb, sum2b, mxb, mnb, cntb, ldb);
    dim3 gP(Nv / 16, Fv / 16);
    pna_post_k<<<gP, 32, 0, stream>>>(x_gnn, sumb, (float*)mxb, (float*)mnb, sum2b, ldb,
                                      Wplp, bplb, convb);
    bn_stats_k<<<Fv, 256, 0, stream>>>(convb, bnm, bnrv);
    gnn_update_k<<<(unsigned)((NF + 255) / 256), 256, 0, stream>>>(x_gnn, convb, bnm, bnrv,
                                                                   bng, bnb, out_xg);

    // --- fused edge-update MLP ---
    edge_mlp_k<<<Ev / 16, 32, 0, stream>>>(out_xg, ea, src, dst, Weu1p, beu1, Weu2p, beu2, out_ea);
}
